// ModelSTAGIN_52226802319572
// MI455X (gfx1250) — compile-verified, matrix operand order checked by reading, not run on hardware
//
#include <hip/hip_runtime.h>
#include <hip/hip_bf16.h>
#include <math.h>

// ---------------- model constants ----------------
namespace {
constexpr int LL    = 4;
constexpr int HDIM  = 256;   // channel dim (multiple of 64)
constexpr int NNODE = 111;   // nodes
constexpr int NP    = 128;   // padded node / 111-col stride
constexpr int TT    = 2048;  // time steps
constexpr int NHEAD = 3;
constexpr int FFND  = 512;
constexpr int HD    = 37;    // head dim
constexpr int QS    = 576;   // qkv padded row stride: 3 segs * 3 heads * 64
constexpr int KSEL  = 8624;  // floor(70*(111*111-1)/100); interp weight exactly 0
constexpr int MROWS = TT * NNODE;  // 227328 (multiple of 64)
}

typedef __attribute__((ext_vector_type(16))) _Float16 v16h;
typedef __attribute__((ext_vector_type(8)))  _Float16 v8h;
typedef __attribute__((ext_vector_type(8)))  float    v8f;
typedef __attribute__((ext_vector_type(4)))  int      v4i;

union AF { v16h v; v8h h8[2]; _Float16 h[16]; };
union CF { v8f  v; float f[8]; };

// k index inside a 16x32 f16 A/B fragment for element e (0..15), lane half kh.
__device__ __forceinline__ int wk_map(int e, int kh) {
  return (e < 8) ? (8 * kh + e) : (16 + 8 * kh + (e - 8));
}

// ---------------- async global->LDS copy (gfx1250), guarded probe ----------------
#if defined(__gfx1250__) && __has_builtin(__builtin_amdgcn_global_load_async_to_lds_b128) && \
    __has_builtin(__builtin_amdgcn_s_wait_asynccnt)
#define HAVE_ASYNC_LDS 1
#else
#define HAVE_ASYNC_LDS 0
#endif

__device__ __forceinline__ void copy8h_to_lds(_Float16* dst, const _Float16* src) {
#if HAVE_ASYNC_LDS
  __builtin_amdgcn_global_load_async_to_lds_b128((v4i*)(src), (v4i*)(dst), 0, 0);
#else
  *(v8h*)dst = *(const v8h*)src;
#endif
}

__device__ __forceinline__ void wait_async_copies() {
#if HAVE_ASYNC_LDS
  __builtin_amdgcn_s_wait_asynccnt(0);
#endif
}

// ---------------- percentile threshold (radix select over 12321 elems) ----------------
__global__ __launch_bounds__(256) void percentile_thr(const float* __restrict__ a,
                                                      float* __restrict__ thr) {
  const int t = blockIdx.x;
  const float* at = a + (size_t)t * NNODE * NNODE;
  const int NNE = NNODE * NNODE;
  __shared__ unsigned hist[256];
  __shared__ unsigned sPrefix;
  __shared__ int sK;
  if (threadIdx.x == 0) { sPrefix = 0u; sK = KSEL; }
  __syncthreads();
  for (int shift = 24; shift >= 0; shift -= 8) {
    hist[threadIdx.x] = 0u;
    __syncthreads();
    const unsigned maskHigh = (shift == 24) ? 0u : (0xFFFFFFFFu << (shift + 8));
    const unsigned pref = sPrefix;
    for (int i = threadIdx.x; i < NNE; i += 256) {
      unsigned u = __float_as_uint(at[i]);
      unsigned key = (u & 0x80000000u) ? ~u : (u | 0x80000000u);
      if ((key & maskHigh) == pref) atomicAdd(&hist[(key >> shift) & 255], 1u);
    }
    __syncthreads();
    if (threadIdx.x == 0) {
      int k = sK; unsigned cum = 0; int d = 0;
      for (; d < 256; d++) { unsigned c = hist[d]; if (cum + c > (unsigned)k) break; cum += c; }
      sK = k - (int)cum;
      sPrefix = pref | ((unsigned)d << shift);
    }
    __syncthreads();
  }
  if (threadIdx.x == 0) {
    unsigned key = sPrefix;
    unsigned u = (key & 0x80000000u) ? (key & 0x7FFFFFFFu) : ~key;
    thr[t] = __uint_as_float(u);
  }
}

// ---------------- f32 -> f16 padded-row conversion: dst(N, Kp) from src(N, K) ----------------
__global__ __launch_bounds__(256) void cvt_pad16(const float* __restrict__ src,
                                                 _Float16* __restrict__ dst,
                                                 long total, int K, int Kp) {
  long i = (long)blockIdx.x * 256 + threadIdx.x;
  if (i >= total) return;
  long n = i / Kp; int k = (int)(i % Kp);
  dst[i] = (k < K) ? (_Float16)src[n * K + k] : (_Float16)0.f;
}

// ---------------- build 0/1 mask tensor (T,128,128) f16, pads zero ----------------
__global__ __launch_bounds__(256) void build_mask(const float* __restrict__ a,
                                                  const float* __restrict__ thr,
                                                  _Float16* __restrict__ m16) {
  long i = (long)blockIdx.x * 256 + threadIdx.x;
  if (i >= (long)TT * NP * NP) return;
  int t = (int)(i >> 14);
  int r = (int)((i >> 7) & 127);
  int c = (int)(i & 127);
  float v = 0.f;
  if (r < NNODE && c < NNODE)
    v = (a[(size_t)t * NNODE * NNODE + r * NNODE + c] > thr[t]) ? 1.f : 0.f;
  m16[i] = (_Float16)v;
}

// ---------------- WMMA GEMM, all-f16 direct-from-global fragments ----------------
// C(M, Cs) = act(A(M, Kp) @ W(Nout, Kp)^T + bias).  M % 64 == 0, Kp % 32 == 0.
// Grid: x = n-tiles (fast, shares A rows in L2), y = m-tiles.
// act: 0 none, 1 relu, 2 gelu(exact), 3 sigmoid, 4 qkv head-pad scatter (Cs=QS).
__global__ __launch_bounds__(128) void gemm16(const _Float16* __restrict__ A, int Kp,
                                              const _Float16* __restrict__ W,
                                              const float* __restrict__ bias,
                                              _Float16* __restrict__ C,
                                              int Nout, int Cs, int act) {
  const int n0 = blockIdx.x * 64;
  const int m0 = blockIdx.y * 64;
  const int lane = threadIdx.x & 31, wave = threadIdx.x >> 5;
  const int nc = lane & 15, kh = lane >> 4;

  const _Float16* Arow = A + (size_t)(m0 + (wave << 4) + nc) * Kp;
  const _Float16* Wr0 = W + (size_t)(n0 + nc) * Kp;
  const _Float16* Wr1 = Wr0 + (size_t)16 * Kp;
  const _Float16* Wr2 = Wr0 + (size_t)32 * Kp;
  const _Float16* Wr3 = Wr0 + (size_t)48 * Kp;

  CF acc[4];
#pragma unroll
  for (int s = 0; s < 4; s++)
#pragma unroll
    for (int r = 0; r < 8; r++) acc[s].f[r] = 0.f;

  for (int k0 = 0; k0 < Kp; k0 += 32) {
    __builtin_prefetch(Arow + k0 + 256, 0, 0);  // stream-ahead on the A row
    AF af, b0, b1, b2, b3;
    af.h8[0] = *(const v8h*)(Arow + k0 + 8 * kh);
    af.h8[1] = *(const v8h*)(Arow + k0 + 16 + 8 * kh);
    b0.h8[0] = *(const v8h*)(Wr0 + k0 + 8 * kh);
    b0.h8[1] = *(const v8h*)(Wr0 + k0 + 16 + 8 * kh);
    b1.h8[0] = *(const v8h*)(Wr1 + k0 + 8 * kh);
    b1.h8[1] = *(const v8h*)(Wr1 + k0 + 16 + 8 * kh);
    b2.h8[0] = *(const v8h*)(Wr2 + k0 + 8 * kh);
    b2.h8[1] = *(const v8h*)(Wr2 + k0 + 16 + 8 * kh);
    b3.h8[0] = *(const v8h*)(Wr3 + k0 + 8 * kh);
    b3.h8[1] = *(const v8h*)(Wr3 + k0 + 16 + 8 * kh);
    acc[0].v = __builtin_amdgcn_wmma_f32_16x16x32_f16(false, af.v, false, b0.v, (short)0, acc[0].v, false, false);
    acc[1].v = __builtin_amdgcn_wmma_f32_16x16x32_f16(false, af.v, false, b1.v, (short)0, acc[1].v, false, false);
    acc[2].v = __builtin_amdgcn_wmma_f32_16x16x32_f16(false, af.v, false, b2.v, (short)0, acc[2].v, false, false);
    acc[3].v = __builtin_amdgcn_wmma_f32_16x16x32_f16(false, af.v, false, b3.v, (short)0, acc[3].v, false, false);
  }

#pragma unroll
  for (int s = 0; s < 4; s++) {
    const int ncol = n0 + (s << 4) + nc;
    const bool valid = (ncol < Nout);
    const float bv = valid ? bias[ncol] : 0.f;
    int dcol = ncol;
    if (act == 4 && valid) {       // qkv scatter: n = seg*111 + head*37 + d
      int seg = ncol / NNODE, rem = ncol % NNODE;
      int hh = rem / HD, d = rem % HD;
      dcol = seg * 192 + hh * 64 + d;
    }
#pragma unroll
    for (int r = 0; r < 8; r++) {
      const int grow = m0 + (wave << 4) + r + 8 * kh;
      float y = acc[s].f[r] + bv;
      if (act == 1)      y = fmaxf(y, 0.f);
      else if (act == 2) y = 0.5f * y * (1.f + erff(y * 0.70710678118f));
      else if (act == 3) y = 1.f / (1.f + __expf(-y));
      if (act == 4) {
        if (valid) C[(size_t)grow * Cs + dcol] = (_Float16)y;  // pads pre-zeroed
      } else {
        C[(size_t)grow * Cs + dcol] = valid ? (_Float16)y : (_Float16)0.f;
      }
    }
  }
}

// ---------------- GIN aggregate: out = mask @ h + eps*h (per graph t) ----------------
// Grid: x = n-tiles(4), y = m-tiles(2), z = t (slow) -> per-graph tiles share cache.
__global__ __launch_bounds__(128) void agg_wmma(const _Float16* __restrict__ m16,
                                                const _Float16* __restrict__ h,
                                                const float* __restrict__ epsp,
                                                _Float16* __restrict__ out) {
  const int n0 = blockIdx.x * 64;
  const int m0 = blockIdx.y * 64;
  const int t  = blockIdx.z;
  const float eps = epsp[0];
  const _Float16* mt = m16 + (size_t)t * NP * NP;
  const _Float16* ht = h + (size_t)t * NNODE * HDIM;
  _Float16* ot = out + (size_t)t * NNODE * HDIM;
  const int lane = threadIdx.x & 31, wave = threadIdx.x >> 5;
  const int nc = lane & 15, kh = lane >> 4;
  __shared__ _Float16 sB[32][72];  // [node-k][channel], 16B-aligned rows

  const _Float16* Arow = mt + (size_t)(m0 + (wave << 4) + nc) * NP;
  CF acc[4];
#pragma unroll
  for (int s = 0; s < 4; s++)
#pragma unroll
    for (int r = 0; r < 8; r++) acc[s].f[r] = 0.f;

  for (int k0 = 0; k0 < NP; k0 += 32) {
    // stage h^T chunk: 32 nodes x 64 channels (async global->LDS when available)
#pragma unroll
    for (int p = 0; p < 2; p++) {
      const int lin = threadIdx.x + p * 128;
      const int kk = lin >> 3, cb = (lin & 7) * 8;
      const int node = k0 + kk;
      if (node < NNODE) {
        copy8h_to_lds(&sB[kk][cb], ht + (size_t)node * HDIM + n0 + cb);
      } else {
        *(v8h*)(&sB[kk][cb]) = (v8h){0, 0, 0, 0, 0, 0, 0, 0};
      }
    }
    wait_async_copies();
    __syncthreads();
    AF af;
    af.h8[0] = *(const v8h*)(Arow + k0 + 8 * kh);
    af.h8[1] = *(const v8h*)(Arow + k0 + 16 + 8 * kh);
#pragma unroll
    for (int s = 0; s < 4; s++) {
      AF bf;
      const int ncol = (s << 4) + nc;
#pragma unroll
      for (int e = 0; e < 16; e++) bf.h[e] = sB[wk_map(e, kh)][ncol];
      acc[s].v = __builtin_amdgcn_wmma_f32_16x16x32_f16(false, af.v, false, bf.v,
                                                        (short)0, acc[s].v, false, false);
    }
    __syncthreads();
  }
#pragma unroll
  for (int s = 0; s < 4; s++) {
    const int ncol = n0 + (s << 4) + nc;
#pragma unroll
    for (int r = 0; r < 8; r++) {
      const int grow = m0 + (wave << 4) + r + 8 * kh;
      if (grow < NNODE) {
        float hv = (float)ht[(size_t)grow * HDIM + ncol];
        ot[(size_t)grow * HDIM + ncol] = (_Float16)(acc[s].f[r] + eps * hv);
      }
    }
  }
}

// ---------------- BatchNorm stats + apply (f16 data, f32 math) ----------------
__global__ __launch_bounds__(256) void col_stats(const _Float16* __restrict__ X, int M, int C,
                                                 float* __restrict__ stats) {
  const int r0 = blockIdx.x * 128;
  int rend = r0 + 128; if (rend > M) rend = M;
  for (int c = threadIdx.x; c < C; c += blockDim.x) {
    float s = 0.f, q = 0.f;
    for (int r = r0; r < rend; r++) {
      float x = (float)X[(size_t)r * C + c];
      s += x; q += x * x;
    }
    atomicAdd(&stats[c], s);
    atomicAdd(&stats[C + c], q);
  }
}

__global__ __launch_bounds__(256) void bn_act(_Float16* __restrict__ X, int M, int C,
                                              const float* __restrict__ stats,
                                              const float* __restrict__ g,
                                              const float* __restrict__ be, int act) {
  const size_t i = (size_t)blockIdx.x * blockDim.x + threadIdx.x;
  if (i >= (size_t)M * C) return;
  const int c = (int)(i % C);
  const float invM = 1.f / (float)M;
  const float mean = stats[c] * invM;
  const float var = stats[C + c] * invM - mean * mean;
  float y = ((float)X[i] - mean) * rsqrtf(var + 1e-5f) * g[c] + be[c];
  if (act == 1)      y = fmaxf(y, 0.f);
  else if (act == 2) y = 0.5f * y * (1.f + erff(y * 0.70710678118f));
  X[i] = (_Float16)y;
}

// ---------------- SERO helpers ----------------
__global__ __launch_bounds__(HDIM) void mean_nodes(const _Float16* __restrict__ h,
                                                   _Float16* __restrict__ xr) {
  const int t = blockIdx.x, c = threadIdx.x;
  const _Float16* ht = h + (size_t)t * NNODE * HDIM;
  float s = 0.f;
  for (int n = 0; n < NNODE; n++) s += (float)ht[(size_t)n * HDIM + c];
  xr[(size_t)t * HDIM + c] = (_Float16)(s * (1.f / (float)NNODE));
}

// hread(t, NP) = attn(t,n) * mean_H h(t,n,:) ; pads written zero
__global__ __launch_bounds__(256) void sero_readout(const _Float16* __restrict__ h,
                                                    const _Float16* __restrict__ attn,
                                                    _Float16* __restrict__ hread) {
  const int row = blockIdx.x * 8 + (threadIdx.x >> 5);  // row over T*NP
  if (row >= TT * NP) return;
  const int lane = threadIdx.x & 31;
  const int t = row >> 7, n = row & 127;
  if (n >= NNODE) { if (lane == 0) hread[row] = (_Float16)0.f; return; }
  const _Float16* hr = h + (size_t)(t * NNODE + n) * HDIM;
  float s = 0.f;
  for (int c = lane; c < HDIM; c += 32) s += (float)hr[c];
  for (int o = 16; o >= 1; o >>= 1) s += __shfl_xor(s, o, 32);
  if (lane == 0)
    hread[row] = (_Float16)((float)attn[(size_t)t * NP + n] * (s / (float)HDIM));
}

// ---------------- flash attention over time (3 heads, hd=37, head-padded qkv) ----------------
__global__ __launch_bounds__(128) void attn_flash(const _Float16* __restrict__ qkv,
                                                  _Float16* __restrict__ o) {
  const int head = blockIdx.x;
  const int qt = blockIdx.y;
  const int lane = threadIdx.x & 31, wave = threadIdx.x >> 5;
  const int nc = lane & 15, kh = lane >> 4;
  const int qrow0 = qt * 64 + wave * 16;
  const int qoff = head * 64, koff = 192 + head * 64, voff = 384 + head * 64;
  __shared__ _Float16 sV[16][72];
  __shared__ _Float16 sP[4][16][32];

  for (int i = lane; i < 256; i += 32)
    sP[wave][i >> 4][16 + (i & 15)] = (_Float16)0.f;  // zero k>=16 half (avoid NaN)

  AF qf[2];
  {
    const _Float16* qrow = qkv + (size_t)(qrow0 + nc) * QS + qoff;
#pragma unroll
    for (int c = 0; c < 2; c++) {
      qf[c].h8[0] = *(const v8h*)(qrow + c * 32 + 8 * kh);
      qf[c].h8[1] = *(const v8h*)(qrow + c * 32 + 16 + 8 * kh);
    }
  }
  CF oacc[4];
#pragma unroll
  for (int s = 0; s < 4; s++)
#pragma unroll
    for (int r = 0; r < 8; r++) oacc[s].f[r] = 0.f;
  float mrow[8], lrow[8];
#pragma unroll
  for (int r = 0; r < 8; r++) { mrow[r] = -INFINITY; lrow[r] = 0.f; }
  const float scale = 0.16439898730f;  // 1/sqrt(37)

  for (int j = 0; j < TT; j += 16) {
    // stage V tile (16 keys x 64 dims)
    {
      const int kk = threadIdx.x >> 3, cb = (threadIdx.x & 7) * 8;
      copy8h_to_lds(&sV[kk][cb], qkv + (size_t)(j + kk) * QS + voff + cb);
    }
    wait_async_copies();
    __syncthreads();

    // S = Q @ K^T : K fragments direct from global (zero-padded dims)
    CF s;
#pragma unroll
    for (int r = 0; r < 8; r++) s.f[r] = 0.f;
    const _Float16* krow = qkv + (size_t)(j + nc) * QS + koff;
#pragma unroll
    for (int c = 0; c < 2; c++) {
      AF bf;
      bf.h8[0] = *(const v8h*)(krow + c * 32 + 8 * kh);
      bf.h8[1] = *(const v8h*)(krow + c * 32 + 16 + 8 * kh);
      s.v = __builtin_amdgcn_wmma_f32_16x16x32_f16(false, qf[c].v, false, bf.v,
                                                   (short)0, s.v, false, false);
    }
    // online softmax; acc element r is row (r + 8*kh), col nc
#pragma unroll
    for (int r = 0; r < 8; r++) {
      float x = s.f[r] * scale;
      float mx = x;
      for (int o2 = 8; o2 >= 1; o2 >>= 1) mx = fmaxf(mx, __shfl_xor(mx, o2, 16));
      const float mnew = fmaxf(mrow[r], mx);
      const float p = __expf(x - mnew);
      float ssum = p;
      for (int o2 = 8; o2 >= 1; o2 >>= 1) ssum += __shfl_xor(ssum, o2, 16);
      const float corr = __expf(mrow[r] - mnew);
      lrow[r] = lrow[r] * corr + ssum;
      mrow[r] = mnew;
#pragma unroll
      for (int t2 = 0; t2 < 4; t2++) oacc[t2].f[r] *= corr;
      sP[wave][r + 8 * kh][nc] = (_Float16)p;
    }
    __syncthreads();
    // O += P @ V
    {
      AF pf;
      pf.h8[0] = *(const v8h*)(&sP[wave][nc][8 * kh]);
      pf.h8[1] = *(const v8h*)(&sP[wave][nc][16 + 8 * kh]);
#pragma unroll
      for (int t2 = 0; t2 < 4; t2++) {
        AF bf;
        const int d = (t2 << 4) + nc;
#pragma unroll
        for (int e = 0; e < 16; e++) {
          const int kk = wk_map(e, kh);
          bf.h[e] = (kk < 16) ? sV[kk][d] : (_Float16)0.f;
        }
        oacc[t2].v = __builtin_amdgcn_wmma_f32_16x16x32_f16(false, pf.v, false, bf.v,
                                                            (short)0, oacc[t2].v, false, false);
      }
    }
    __syncthreads();
  }
#pragma unroll
  for (int t2 = 0; t2 < 4; t2++) {
    const int d = (t2 << 4) + nc;
    if (d >= HD) continue;
#pragma unroll
    for (int r = 0; r < 8; r++) {
      const int qr = qrow0 + r + 8 * kh;
      o[(size_t)qr * NP + head * HD + d] = (_Float16)(oacc[t2].f[r] / lrow[r]);
    }
  }
}

// ---------------- LayerNorm rows (width 111, stride NP), optional residual ----------------
__global__ __launch_bounds__(256) void ln_rows(const _Float16* __restrict__ X,
                                               const _Float16* __restrict__ R,
                                               const float* __restrict__ g,
                                               const float* __restrict__ b,
                                               _Float16* __restrict__ Y) {
  const int row = blockIdx.x * 8 + (threadIdx.x >> 5);
  if (row >= TT) return;
  const int lane = threadIdx.x & 31;
  const _Float16* xp = X + (size_t)row * NP;
  const _Float16* rp = R ? (R + (size_t)row * NP) : nullptr;
  float s = 0.f;
  for (int c = lane; c < NNODE; c += 32) {
    float v = (float)xp[c]; if (rp) v += (float)rp[c]; s += v;
  }
  for (int o = 16; o >= 1; o >>= 1) s += __shfl_xor(s, o, 32);
  const float mean = s / (float)NNODE;
  float q = 0.f;
  for (int c = lane; c < NNODE; c += 32) {
    float v = (float)xp[c]; if (rp) v += (float)rp[c];
    float d = v - mean; q += d * d;
  }
  for (int o = 16; o >= 1; o >>= 1) q += __shfl_xor(q, o, 32);
  const float inv = rsqrtf(q / (float)NNODE + 1e-5f);
  for (int c = lane; c < NP; c += 32) {
    float y = 0.f;
    if (c < NNODE) {
      float v = (float)xp[c]; if (rp) v += (float)rp[c];
      y = (v - mean) * inv * g[c] + b[c];
    }
    Y[(size_t)row * NP + c] = (_Float16)y;
  }
}

// ---------------- output conversion ----------------
__global__ __launch_bounds__(256) void transpose_logit(const _Float16* __restrict__ h,
                                                       float* __restrict__ out) {
  const int i = blockIdx.x * 256 + threadIdx.x;
  if (i >= NNODE * TT) return;
  const int n = i / TT, t = i % TT;
  out[i] = (float)h[(size_t)t * NP + n];
}

__global__ __launch_bounds__(256) void cvt_out(const _Float16* __restrict__ h,
                                               float* __restrict__ out, long total) {
  long i = (long)blockIdx.x * 256 + threadIdx.x;
  if (i >= total) return;
  out[i] = (float)h[i];
}

// ---------------- host-side orchestration ----------------
static inline void run_gemm(const _Float16* A, int Kp, const _Float16* W, const float* b,
                            _Float16* C, int M, int Nout, int Cs, int act, hipStream_t s) {
  dim3 g((Nout + 63) / 64, M / 64);  // n-tiles fastest: A rows shared in L2
  gemm16<<<g, 128, 0, s>>>(A, Kp, W, b, C, Nout, Cs, act);
}

static inline void run_bn(_Float16* X, int M, int C, float* stats,
                          const float* g, const float* be, int act, hipStream_t s) {
  (void)hipMemsetAsync(stats, 0, (size_t)2 * C * sizeof(float), s);
  col_stats<<<(M + 127) / 128, 256, 0, s>>>(X, M, C, stats);
  size_t tot = (size_t)M * C;
  bn_act<<<(unsigned)((tot + 255) / 256), 256, 0, s>>>(X, M, C, stats, g, be, act);
}

static inline void run_cvt(const float* src, _Float16* dst, long rows, int K, int Kp,
                           hipStream_t s) {
  long tot = rows * Kp;
  cvt_pad16<<<(unsigned)((tot + 255) / 256), 256, 0, s>>>(src, dst, tot, K, Kp);
}

extern "C" void kernel_launch(void* const* d_in, const int* in_sizes, int n_in,
                              void* d_out, int out_size, void* d_ws, size_t ws_size,
                              hipStream_t stream) {
  (void)in_sizes; (void)n_in; (void)out_size; (void)ws_size;
  const float* a       = (const float*)d_in[0];
  const float* init_w  = (const float*)d_in[1];
  const float* init_b  = (const float*)d_in[2];
  const float* gin_eps = (const float*)d_in[3];
  const float* gin_w1  = (const float*)d_in[4];
  const float* gin_b1  = (const float*)d_in[5];
  const float* gin_g1  = (const float*)d_in[6];
  const float* gin_be1 = (const float*)d_in[7];
  const float* gin_w2  = (const float*)d_in[8];
  const float* gin_b2  = (const float*)d_in[9];
  const float* gin_g2  = (const float*)d_in[10];
  const float* gin_be2 = (const float*)d_in[11];
  const float* sero_ew = (const float*)d_in[12];
  const float* sero_eb = (const float*)d_in[13];
  const float* sero_g  = (const float*)d_in[14];
  const float* sero_be = (const float*)d_in[15];
  const float* sero_aw = (const float*)d_in[16];
  const float* sero_ab = (const float*)d_in[17];
  const float* tr_inw  = (const float*)d_in[18];
  const float* tr_inb  = (const float*)d_in[19];
  const float* tr_ow   = (const float*)d_in[20];
  const float* tr_ob   = (const float*)d_in[21];
  const float* tr_ln1g = (const float*)d_in[22];
  const float* tr_ln1b = (const float*)d_in[23];
  const float* tr_ln2g = (const float*)d_in[24];
  const float* tr_ln2b = (const float*)d_in[25];
  const float* tr_m1w  = (const float*)d_in[26];
  const float* tr_m1b  = (const float*)d_in[27];
  const float* tr_m2w  = (const float*)d_in[28];
  const float* tr_m2b  = (const float*)d_in[29];

  char* ws = (char*)d_ws;
  size_t off = 0;
  auto takeB = [&](size_t bytes) -> char* {
    char* p = ws + off;
    off = (off + bytes + 255) & ~(size_t)255;
    return p;
  };
  auto takeH = [&](size_t n) -> _Float16* { return (_Float16*)takeB(n * 2); };
  auto takeF = [&](size_t n) -> float* { return (float*)takeB(n * 4); };

  float* thr   = takeF(TT);
  float* stats = takeF(2 * 512);
  _Float16* a16    = takeH((size_t)MROWS * NP);
  _Float16* mask16 = takeH((size_t)TT * NP * NP);
  _Float16* bufA   = takeH((size_t)MROWS * HDIM);
  _Float16* bufB   = takeH((size_t)MROWS * HDIM);
  _Float16* xr16   = takeH((size_t)TT * HDIM);
  _Float16* xe16   = takeH((size_t)TT * HDIM);
  _Float16* attnB  = takeH((size_t)TT * NP);
  _Float16* hread  = takeH((size_t)TT * NP);
  _Float16* qkv16  = takeH((size_t)TT * QS);
  _Float16* oattn  = takeH((size_t)TT * NP);
  _Float16* otmp   = takeH((size_t)TT * NP);
  _Float16* oln    = takeH((size_t)TT * NP);
  _Float16* m1B    = takeH((size_t)TT * FFND);
  _Float16* moutB  = takeH((size_t)TT * NP);
  _Float16* hatt   = takeH((size_t)TT * NP);
  _Float16* w16_init = takeH((size_t)HDIM * NP);
  _Float16* w16_g1   = takeH((size_t)LL * HDIM * HDIM);
  _Float16* w16_g2   = takeH((size_t)LL * HDIM * HDIM);
  _Float16* w16_ew   = takeH((size_t)LL * HDIM * HDIM);
  _Float16* w16_aw   = takeH((size_t)LL * NNODE * HDIM);
  _Float16* w16_inw  = takeH((size_t)LL * 3 * NNODE * NP);
  _Float16* w16_ow   = takeH((size_t)LL * NNODE * NP);
  _Float16* w16_m1   = takeH((size_t)LL * FFND * NP);
  _Float16* w16_m2   = takeH((size_t)LL * NNODE * FFND);

  // --- one-time conversions ---
  percentile_thr<<<TT, 256, 0, stream>>>(a, thr);
  run_cvt(a, a16, MROWS, NNODE, NP, stream);
  build_mask<<<(unsigned)(((long)TT * NP * NP + 255) / 256), 256, 0, stream>>>(a, thr, mask16);
  run_cvt(init_w, w16_init, HDIM, NNODE, NP, stream);
  run_cvt(gin_w1, w16_g1, (long)LL * HDIM, HDIM, HDIM, stream);
  run_cvt(gin_w2, w16_g2, (long)LL * HDIM, HDIM, HDIM, stream);
  run_cvt(sero_ew, w16_ew, (long)LL * HDIM, HDIM, HDIM, stream);
  run_cvt(sero_aw, w16_aw, (long)LL * NNODE, HDIM, HDIM, stream);
  run_cvt(tr_inw, w16_inw, (long)LL * 3 * NNODE, NNODE, NP, stream);
  run_cvt(tr_ow, w16_ow, (long)LL * NNODE, NNODE, NP, stream);
  run_cvt(tr_m1w, w16_m1, (long)LL * FFND, NNODE, NP, stream);
  run_cvt(tr_m2w, w16_m2, (long)LL * NNODE, FFND, FFND, stream);

  // --- initial embedding: h = a @ init_w^T + init_b ---
  _Float16* cur = bufA;
  _Float16* oth = bufB;
  run_gemm(a16, NP, w16_init, init_b, cur, MROWS, HDIM, HDIM, 0, stream);

  for (int l = 0; l < LL; l++) {
    // GIN aggregate + MLP with BatchNorm
    agg_wmma<<<dim3(4, 2, TT), 128, 0, stream>>>(mask16, cur, gin_eps + l, oth);
    run_gemm(oth, HDIM, w16_g1 + (size_t)l * HDIM * HDIM, gin_b1 + (size_t)l * HDIM,
             cur, MROWS, HDIM, HDIM, 0, stream);
    run_bn(cur, MROWS, HDIM, stats, gin_g1 + (size_t)l * HDIM, gin_be1 + (size_t)l * HDIM, 1, stream);
    run_gemm(cur, HDIM, w16_g2 + (size_t)l * HDIM * HDIM, gin_b2 + (size_t)l * HDIM,
             oth, MROWS, HDIM, HDIM, 0, stream);
    run_bn(oth, MROWS, HDIM, stats, gin_g2 + (size_t)l * HDIM, gin_be2 + (size_t)l * HDIM, 1, stream);
    { _Float16* tmp = cur; cur = oth; oth = tmp; }  // cur == h_bridge

    // SERO readout
    mean_nodes<<<TT, HDIM, 0, stream>>>(cur, xr16);
    run_gemm(xr16, HDIM, w16_ew + (size_t)l * HDIM * HDIM, sero_eb + (size_t)l * HDIM,
             xe16, TT, HDIM, HDIM, 0, stream);
    run_bn(xe16, TT, HDIM, stats, sero_g + (size_t)l * HDIM, sero_be + (size_t)l * HDIM, 2, stream);
    run_gemm(xe16, HDIM, w16_aw + (size_t)l * NNODE * HDIM, sero_ab + (size_t)l * NNODE,
             attnB, TT, NNODE, NP, 3, stream);
    sero_readout<<<(TT * NP + 7) / 8, 256, 0, stream>>>(cur, attnB, hread);

    // transformer over time
    (void)hipMemsetAsync(qkv16, 0, (size_t)TT * QS * 2, stream);
    run_gemm(hread, NP, w16_inw + (size_t)l * 3 * NNODE * NP, tr_inb + (size_t)l * 3 * NNODE,
             qkv16, TT, 3 * NNODE, QS, 4, stream);
    (void)hipMemsetAsync(oattn, 0, (size_t)TT * NP * 2, stream);
    attn_flash<<<dim3(NHEAD, TT / 64), 128, 0, stream>>>(qkv16, oattn);
    run_gemm(oattn, NP, w16_ow + (size_t)l * NNODE * NP, tr_ob + (size_t)l * NNODE,
             otmp, TT, NNODE, NP, 0, stream);
    ln_rows<<<(TT + 7) / 8, 256, 0, stream>>>(otmp, nullptr,
             tr_ln1g + (size_t)l * NNODE, tr_ln1b + (size_t)l * NNODE, oln);
    run_gemm(oln, NP, w16_m1 + (size_t)l * FFND * NP, tr_m1b + (size_t)l * FFND,
             m1B, TT, FFND, FFND, 1, stream);
    run_gemm(m1B, FFND, w16_m2 + (size_t)l * NNODE * FFND, tr_m2b + (size_t)l * NNODE,
             moutB, TT, NNODE, NP, 0, stream);
    ln_rows<<<(TT + 7) / 8, 256, 0, stream>>>(oln, moutB,
             tr_ln2g + (size_t)l * NNODE, tr_ln2b + (size_t)l * NNODE, hatt);
  }

  // outputs: logit (N,T) f32, then h_bridge[:,0] == final h (T,N,H) f32
  transpose_logit<<<(NNODE * TT + 255) / 256, 256, 0, stream>>>(hatt, (float*)d_out);
  cvt_out<<<(unsigned)(((long)MROWS * HDIM + 255) / 256), 256, 0, stream>>>(
      cur, (float*)d_out + (size_t)NNODE * TT, (long)MROWS * HDIM);
}
